// FullMHLA_14147622273639
// MI455X (gfx1250) — matrix operand order repaired and checked by reading, not compiled, not run on hardware
//
#include <hip/hip_runtime.h>

// ---------------------------------------------------------------------------
// MLA forward (DeepSeek-style) for MI455X / gfx1250, wave32 + WMMA f16 + TDM.
// ---------------------------------------------------------------------------

typedef _Float16 half_t;
typedef __attribute__((ext_vector_type(16))) _Float16 v16h;
typedef __attribute__((ext_vector_type(8)))  float    v8f;
typedef __attribute__((ext_vector_type(4)))  unsigned int u32v4;

#define B_    2
#define T_    2048
#define C_    2048
#define NH_   16
#define HS_   128
#define NLQ_  512
#define NLKV_ 512
#define DHR_  64
#define DQK   576      // NLKV + DHR (concatenated latent + rope dims)
#define DV    512      // value latent width

// ---------------- Tensor Data Mover (async tensor load -> LDS) -------------
#if defined(__has_builtin)
#if __has_builtin(__builtin_amdgcn_tensor_load_to_lds) && \
    __has_builtin(__builtin_amdgcn_s_wait_tensorcnt)
#define USE_TDM 1
#endif
#endif
#ifndef USE_TDM
#define USE_TDM 0
#endif

#if USE_TDM
typedef __attribute__((ext_vector_type(4))) unsigned int tdmu4;
typedef __attribute__((ext_vector_type(8))) int          tdmi8;
typedef __attribute__((ext_vector_type(4))) int          tdmi4;

// 2D tile load: tile_y rows of tile_x elements (2B each), global row stride
// row_stride_elems, packed contiguously into LDS at lds_off (optional LDS
// padding via extra_flags = pad_enable/pad_interval/pad_amount bits of G1.d0).
__device__ __forceinline__ void tdm_load_2d(unsigned lds_off,
                                            const half_t* gptr,
                                            unsigned tile_x, unsigned tile_y,
                                            unsigned long long row_stride_elems,
                                            unsigned extra_flags) {
  const unsigned long long ga = reinterpret_cast<unsigned long long>(gptr);
  tdmu4 g0;
  g0[0] = 1u;                                            // count=1 (valid), user mode
  g0[1] = lds_off;                                       // LDS byte address
  g0[2] = (unsigned)(ga & 0xFFFFFFFFull);                // global_addr[31:0]
  g0[3] = (unsigned)((ga >> 32) & 0x01FFFFFFull)         // global_addr[56:32]
          | (2u << 30);                                  // type = 2 ("image")
  const unsigned td = 1u << 22;                          // generous tensor dims
  tdmi8 g1;
  g1[0] = (int)((1u << 16) | extra_flags);               // data_size=2B (+pad)
  g1[1] = (int)((td & 0xFFFFu) << 16);                   // tensor_dim0 lo16
  g1[2] = (int)(((td >> 16) & 0xFFFFu) | ((td & 0xFFFFu) << 16)); // td0 hi | td1 lo
  g1[3] = (int)(((td >> 16) & 0xFFFFu) | (tile_x << 16));         // td1 hi | tile_dim0
  g1[4] = (int)(tile_y & 0xFFFFu);                       // tile_dim1 (tile_dim2=0)
  g1[5] = (int)(unsigned)(row_stride_elems & 0xFFFFFFFFull);      // dim0 stride lo
  g1[6] = (int)(unsigned)((row_stride_elems >> 32) & 0xFFFFull);  // stride hi
  g1[7] = 0;
  tdmi4 gz = {};
#if __clang_major__ >= 23
  tdmi8 z8 = {};
  __builtin_amdgcn_tensor_load_to_lds(g0, g1, gz, gz, z8, 0);
#else
  __builtin_amdgcn_tensor_load_to_lds(g0, g1, gz, gz, 0);
#endif
}
#endif  // USE_TDM

// ---------------- WMMA fragment helpers (16x16x32 f16 -> f32) --------------
__device__ __forceinline__ v16h ld_afrag(const half_t* base, int stride,
                                         int row0, int k0, int lane) {
  const half_t* p = base + (long long)(row0 + (lane & 15)) * stride
                         + k0 + ((lane >> 4) << 3);
  v16h a;
  ((u32v4*)&a)[0] = *(const u32v4*)(p);
  ((u32v4*)&a)[1] = *(const u32v4*)(p + 16);
  return a;
}
__device__ __forceinline__ v16h ld_bfrag(const half_t* base, int stride,
                                         int col0, int k0, int lane) {
  const half_t* p = base + (long long)(col0 + (lane & 15)) * stride
                         + k0 + ((lane >> 4) << 3);
  v16h b;
  ((u32v4*)&b)[0] = *(const u32v4*)(p);
  ((u32v4*)&b)[1] = *(const u32v4*)(p + 16);
  return b;
}
__device__ __forceinline__ v8f wmma32(v16h a, v16h b, v8f c) {
  return __builtin_amdgcn_wmma_f32_16x16x32_f16(false, a, false, b,
                                                (short)0, c, false, false);
}

// ---------------- Generic batched NT GEMM: C[m,n] = sum_k A[m,k]*B[n,k] ----
#define BM 128
#define BN 128
#define BK 32
#define LKP (BK + 8)

template <bool HALF_OUT>
__global__ __launch_bounds__(256)
void gemm_nt(const half_t* __restrict__ A, const half_t* __restrict__ Bm,
             float* __restrict__ Cf, half_t* __restrict__ Ch,
             int M, int N, int K, int lda, int ldb, int ldc,
             long long sA, long long sB, long long sC) {
  __shared__ half_t As[BM][LKP];
  __shared__ half_t Bs[BN][LKP];
  const int z = blockIdx.z;
  A  += (long long)z * sA;
  Bm += (long long)z * sB;
  const long long cbase = (long long)z * sC;
  const int m0   = blockIdx.y * BM;
  const int n0   = blockIdx.x * BN;
  const int tid  = threadIdx.x;
  const int lane = tid & 31;
  const int wid  = tid >> 5;
  const int wr   = (wid & 3) * 32;
  const int wc   = (wid >> 2) * 64;

  v8f acc[2][4];
#pragma unroll
  for (int i = 0; i < 2; ++i)
#pragma unroll
    for (int j = 0; j < 4; ++j) { v8f zz = {}; acc[i][j] = zz; }

  const int ar = tid >> 1;
  const int ac = (tid & 1) * 16;

  for (int k0 = 0; k0 < K; k0 += BK) {
    {
      const half_t* ga = A + (long long)(m0 + ar) * lda + k0 + ac;
      *(u32v4*)&As[ar][ac]     = *(const u32v4*)(ga);
      *(u32v4*)&As[ar][ac + 8] = *(const u32v4*)(ga + 8);
      if (k0 + BK < K) __builtin_prefetch(ga + BK, 0, 0);
    }
    {
      if (n0 + ar < N) {
        const half_t* gb = Bm + (long long)(n0 + ar) * ldb + k0 + ac;
        *(u32v4*)&Bs[ar][ac]     = *(const u32v4*)(gb);
        *(u32v4*)&Bs[ar][ac + 8] = *(const u32v4*)(gb + 8);
        if (k0 + BK < K) __builtin_prefetch(gb + BK, 0, 0);
      } else {
        u32v4 zz = {};
        *(u32v4*)&Bs[ar][ac]     = zz;
        *(u32v4*)&Bs[ar][ac + 8] = zz;
      }
    }
    __syncthreads();
    v16h af[2], bf[4];
#pragma unroll
    for (int i = 0; i < 2; ++i) af[i] = ld_afrag(&As[0][0], LKP, wr + i * 16, 0, lane);
#pragma unroll
    for (int j = 0; j < 4; ++j) bf[j] = ld_bfrag(&Bs[0][0], LKP, wc + j * 16, 0, lane);
#pragma unroll
    for (int i = 0; i < 2; ++i)
#pragma unroll
      for (int j = 0; j < 4; ++j)
        acc[i][j] = wmma32(af[i], bf[j], acc[i][j]);
    __syncthreads();
  }

#pragma unroll
  for (int i = 0; i < 2; ++i) {
    const int rbase = m0 + wr + i * 16 + ((lane >> 4) << 3);
#pragma unroll
    for (int j = 0; j < 4; ++j) {
      const int col = n0 + wc + j * 16 + (lane & 15);
      if (col < N) {
#pragma unroll
        for (int r = 0; r < 8; ++r) {
          const long long idx = cbase + (long long)(rbase + r) * ldc + col;
          if constexpr (HALF_OUT) Ch[idx] = (half_t)acc[i][j][r];
          else                    Cf[idx] = acc[i][j][r];
        }
      }
    }
  }
}

// ---------------- Flash attention over concatenated latents ----------------
// qke: [b][h][t][576] = [qk | rope(q_r)],  ckve: [b][s][576] = [c_kv | k_r]
// ckvT: [b][512][T] = c_kv transposed, ctxh: [b][h][t][512]
#define FT 64
#define FS 64
#define QLD (DQK + 8)   // 584 halfs (padded; plain vector loads)
#define KLD DQK         // 576 halfs (unpadded; TDM writes contiguous rows)
#define VLD (FS + 8)    // 72 halfs  (TDM pad: 4 DWORDs after every 32 DWORDs)

__global__ __launch_bounds__(256)
void mla_flash(const half_t* __restrict__ qke, const half_t* __restrict__ ckve,
               const half_t* __restrict__ ckvT, half_t* __restrict__ ctxh,
               float scale) {
  __shared__ half_t Ql[FT][QLD];
  __shared__ half_t Kl[FS][KLD];
  __shared__ half_t Vl[DV][VLD];
  __shared__ float  Sl[FT][FS + 2];
  __shared__ half_t Pl[FT][VLD];
  __shared__ float  mrow[FT], lrow[FT], crow[FT];

  const int bh = blockIdx.y;          // b*NH + h
  const int b  = bh >> 4;
  const int rt = blockIdx.x;
  const int t0 = rt * FT;
  const half_t* Qg = qke  + ((long long)bh * T_ + t0) * DQK;
  const half_t* Kg = ckve + (long long)b * T_ * DQK;
  const half_t* Vg = ckvT + (long long)b * DV * T_;
  half_t*       Og = ctxh + ((long long)bh * T_ + t0) * DV;

  const int tid  = threadIdx.x;
  const int lane = tid & 31;
  const int wid  = tid >> 5;
  const int rm   = (wid & 3) * 16;    // ctx/P row tile base
  const int cg   = (wid >> 2) * 256;  // ctx col group base
  const int stm  = (wid >> 1) * 16;   // S row tile for this wave
  const int stn  = (wid & 1) * 32;    // S col strip (2 tiles)

  for (int i = tid; i < FT * (DQK / 8); i += 256) {
    int r = i / (DQK / 8), c = (i % (DQK / 8)) * 8;
    *(u32v4*)&Ql[r][c] = *(const u32v4*)&Qg[(long long)r * DQK + c];
  }
  if (tid < FT) { mrow[tid] = -1e30f; lrow[tid] = 0.f; }

  v8f acc[16];
#pragma unroll
  for (int i = 0; i < 16; ++i) { v8f zz = {}; acc[i] = zz; }

  for (int j = 0; j <= rt; ++j) {
    const int s0 = j * FS;
#if USE_TDM
    if (wid == 0) {
      // K tile: 64 rows x 576 halfs, contiguous in global and LDS.
      tdm_load_2d((unsigned)reinterpret_cast<unsigned long long>(&Kl[0][0]),
                  Kg + (long long)s0 * DQK, DQK, FS,
                  (unsigned long long)DQK, 0u);
      // V^T tile: 512 rows x 64 halfs, global row stride T; LDS rows padded
      // to 72 halfs: pad_enable | pad_interval=4 (32 DW) | pad_amount=3 (4 DW).
      tdm_load_2d((unsigned)reinterpret_cast<unsigned long long>(&Vl[0][0]),
                  Vg + s0, FS, DV, (unsigned long long)T_,
                  (1u << 20) | (4u << 22) | (3u << 25));
      __builtin_amdgcn_s_wait_tensorcnt(0);
    }
#else
    for (int i = tid; i < FS * (DQK / 8); i += 256) {
      int r = i / (DQK / 8), c = (i % (DQK / 8)) * 8;
      *(u32v4*)&Kl[r][c] = *(const u32v4*)&Kg[(long long)(s0 + r) * DQK + c];
    }
    for (int i = tid; i < DV * (FS / 8); i += 256) {
      int r = i / (FS / 8), c = (i % (FS / 8)) * 8;
      *(u32v4*)&Vl[r][c] = *(const u32v4*)&Vg[(long long)r * T_ + s0 + c];
    }
#endif
    __syncthreads();

    // S = [qk|qr] @ [ckv|kr]^T  (K = 576), 2 tiles per wave
    v8f sa0 = {}, sa1 = {};
    for (int kk = 0; kk < DQK / 32; ++kk) {
      v16h a  = ld_afrag(&Ql[0][0], QLD, stm, kk * 32, lane);
      v16h b0 = ld_bfrag(&Kl[0][0], KLD, stn,      kk * 32, lane);
      v16h b1 = ld_bfrag(&Kl[0][0], KLD, stn + 16, kk * 32, lane);
      sa0 = wmma32(a, b0, sa0);
      sa1 = wmma32(a, b1, sa1);
    }
#pragma unroll
    for (int jj = 0; jj < 2; ++jj) {
      const int col = stn + jj * 16 + (lane & 15);
      const int sg  = s0 + col;
#pragma unroll
      for (int r = 0; r < 8; ++r) {
        const int row = stm + ((lane >> 4) << 3) + r;
        const float v = (jj ? sa1[r] : sa0[r]) * scale;
        Sl[row][col] = (sg <= t0 + row) ? v : -1e30f;   // causal mask
      }
    }
    __syncthreads();

    // online softmax: one thread per row
    if (tid < FT) {
      float mold = mrow[tid];
      float mx = mold;
      for (int c = 0; c < FS; ++c) mx = fmaxf(mx, Sl[tid][c]);
      float corr = __expf(mold - mx);
      float sum = 0.f;
      for (int c = 0; c < FS; ++c) {
        float pv = __expf(Sl[tid][c] - mx);
        sum += pv;
        Pl[tid][c] = (half_t)pv;
      }
      lrow[tid] = lrow[tid] * corr + sum;
      mrow[tid] = mx;
      crow[tid] = corr;
    }
    __syncthreads();

    // rescale running ctx then accumulate P @ V
    float cf[8];
#pragma unroll
    for (int r = 0; r < 8; ++r) cf[r] = crow[rm + ((lane >> 4) << 3) + r];
#pragma unroll
    for (int ct = 0; ct < 16; ++ct)
#pragma unroll
      for (int r = 0; r < 8; ++r) acc[ct][r] *= cf[r];

#pragma unroll
    for (int kk = 0; kk < 2; ++kk) {
      v16h a = ld_afrag(&Pl[0][0], VLD, rm, kk * 32, lane);
#pragma unroll
      for (int ct = 0; ct < 16; ++ct) {
        v16h bv = ld_bfrag(&Vl[0][0], VLD, cg + ct * 16, kk * 32, lane);
        acc[ct] = wmma32(a, bv, acc[ct]);
      }
    }
    __syncthreads();
  }

  float li[8];
#pragma unroll
  for (int r = 0; r < 8; ++r) li[r] = 1.f / lrow[rm + ((lane >> 4) << 3) + r];
#pragma unroll
  for (int ct = 0; ct < 16; ++ct) {
    const int col = cg + ct * 16 + (lane & 15);
#pragma unroll
    for (int r = 0; r < 8; ++r) {
      const int row = rm + ((lane >> 4) << 3) + r;
      Og[(long long)row * DV + col] = (half_t)(acc[ct][r] * li[r]);
    }
  }
}

// ---------------- small elementwise / layout kernels ----------------------
__global__ void cast_f32_to_f16(const float* __restrict__ in,
                                half_t* __restrict__ out, long long n) {
  long long i = (long long)blockIdx.x * blockDim.x + threadIdx.x;
  if (i < n) out[i] = (half_t)in[i];
}

template <typename TIN>
__global__ __launch_bounds__(256)
void transpose_cast(const TIN* __restrict__ in, half_t* __restrict__ out,
                    int R, int Ccols, int ldin, int ldout,
                    long long sIn, long long sOut) {
  __shared__ half_t tile[32][33];
  const int z = blockIdx.z;
  in  += (long long)z * sIn;
  out += (long long)z * sOut;
  const int r0 = blockIdx.y * 32;
  const int c0 = blockIdx.x * 32;
  const int tx = threadIdx.x & 31;
  const int ty = threadIdx.x >> 5;
  for (int rr = ty; rr < 32; rr += 8) {
    const int r = r0 + rr, c = c0 + tx;
    tile[rr][tx] = (r < R && c < Ccols) ? (half_t)in[(long long)r * ldin + c]
                                        : (half_t)0.f;
  }
  __syncthreads();
  for (int cc = ty; cc < 32; cc += 8) {
    const int orow = c0 + cc, ocol = r0 + tx;
    if (orow < Ccols && ocol < R)
      out[(long long)orow * ldout + ocol] = tile[tx][cc];
  }
}

__global__ void rope_k_kernel(const half_t* __restrict__ ckr,
                              const float* __restrict__ cosp,
                              const float* __restrict__ sinp,
                              half_t* __restrict__ ckve) {
  long long i = (long long)blockIdx.x * blockDim.x + threadIdx.x;
  const long long n = (long long)B_ * T_ * (DHR_ / 2);
  if (i >= n) return;
  const int p = (int)(i & 31);
  const long long bt = i >> 5;
  const int t = (int)(bt & (T_ - 1));
  float re = (float)ckr[bt * DHR_ + 2 * p];
  float im = (float)ckr[bt * DHR_ + 2 * p + 1];
  float c = cosp[t * 32 + p], s = sinp[t * 32 + p];
  half_t* o = ckve + bt * DQK + NLKV_;
  o[2 * p]     = (half_t)(re * c - im * s);
  o[2 * p + 1] = (half_t)(re * s + im * c);
}

__global__ void rope_q_kernel(const half_t* __restrict__ cqr,
                              const float* __restrict__ cosp,
                              const float* __restrict__ sinp,
                              half_t* __restrict__ qke) {
  long long i = (long long)blockIdx.x * blockDim.x + threadIdx.x;
  const long long n = (long long)B_ * T_ * NH_ * (DHR_ / 2);
  if (i >= n) return;
  const int p = (int)(i & 31);
  long long r = i >> 5;
  const int h = (int)(r & (NH_ - 1));
  const long long bt = r >> 4;
  const int t = (int)(bt & (T_ - 1));
  const int b = (int)(bt >> 11);
  float re = (float)cqr[bt * (NH_ * DHR_) + h * DHR_ + 2 * p];
  float im = (float)cqr[bt * (NH_ * DHR_) + h * DHR_ + 2 * p + 1];
  float c = cosp[t * 32 + p], s = sinp[t * 32 + p];
  half_t* o = qke + ((long long)(b * NH_ + h) * T_ + t) * DQK + NLKV_;
  o[2 * p]     = (half_t)(re * c - im * s);
  o[2 * p + 1] = (half_t)(re * s + im * c);
}

// ---------------------------------------------------------------------------
extern "C" void kernel_launch(void* const* d_in, const int* in_sizes, int n_in,
                              void* d_out, int out_size, void* d_ws, size_t ws_size,
                              hipStream_t stream) {
  (void)in_sizes; (void)n_in; (void)out_size; (void)ws_size;
  const float* x    = (const float*)d_in[0];
  const float* cosp = (const float*)d_in[1];
  const float* sinp = (const float*)d_in[2];
  const float* Wdq  = (const float*)d_in[3];
  const float* Wuq  = (const float*)d_in[4];
  const float* Wdkv = (const float*)d_in[5];
  const float* Wuk  = (const float*)d_in[6];
  const float* Wuv  = (const float*)d_in[7];
  const float* Wqr  = (const float*)d_in[8];
  const float* Wkr  = (const float*)d_in[9];
  const float* Wo   = (const float*)d_in[10];
  float* out = (float*)d_out;

  char* wp = (char*)d_ws;
  auto alloc_h = [&](long long nelem) {
    half_t* r = (half_t*)wp;
    wp += ((nelem * 2 + 255) / 256) * 256;
    return r;
  };
  const long long BT = (long long)B_ * T_;
  half_t* xh    = alloc_h(BT * C_);
  half_t* Wdqh  = alloc_h((long long)NLQ_ * C_);
  half_t* Wdkvh = alloc_h((long long)NLKV_ * C_);
  half_t* Wkrh  = alloc_h((long long)DHR_ * C_);
  half_t* Wqrh  = alloc_h((long long)NH_ * DHR_ * NLQ_);
  half_t* Wuqh  = alloc_h((long long)C_ * NLQ_);
  half_t* Woh   = alloc_h((long long)C_ * C_);
  half_t* WuvT  = alloc_h((long long)NLKV_ * C_);           // 512 x 2048
  half_t* WukT  = alloc_h((long long)NH_ * NLKV_ * HS_);    // [16][512][128]
  half_t* cqh   = alloc_h(BT * NLQ_);
  half_t* ckve  = alloc_h(BT * DQK);                        // [b][s][576]
  half_t* ckvT  = alloc_h((long long)B_ * DV * T_);         // [b][512][T]
  half_t* ckrh  = alloc_h(BT * DHR_);
  half_t* cqrh  = alloc_h(BT * NH_ * DHR_);
  half_t* keffT = alloc_h((long long)NH_ * NLKV_ * NLQ_);   // [16][512][512]
  half_t* vpreh = alloc_h((long long)NLKV_ * C_);           // 512 x 2048
  half_t* veffT = alloc_h((long long)C_ * NLKV_);           // 2048 x 512
  half_t* qkeh  = alloc_h((long long)B_ * NH_ * T_ * DQK);  // [b][h][t][576]
  half_t* ctxh  = alloc_h((long long)B_ * NH_ * T_ * DV);   // [b][h][t][512]

  dim3 blk(256);
  auto cast = [&](const float* i, half_t* o, long long n) {
    cast_f32_to_f16<<<dim3((unsigned)((n + 255) / 256)), blk, 0, stream>>>(i, o, n);
  };
  auto gemmH = [&](const half_t* A, const half_t* Bm, half_t* Cout,
                   int M, int N, int K, int lda, int ldb, int ldc,
                   long long sA, long long sB, long long sC, int batch) {
    dim3 g((N + BN - 1) / BN, M / BM, batch);
    gemm_nt<true><<<g, blk, 0, stream>>>(A, Bm, nullptr, Cout, M, N, K,
                                         lda, ldb, ldc, sA, sB, sC);
  };

  // ---- f16 conversions of inputs ----
  cast(x,    xh,    BT * C_);
  cast(Wdq,  Wdqh,  (long long)NLQ_ * C_);
  cast(Wdkv, Wdkvh, (long long)NLKV_ * C_);
  cast(Wkr,  Wkrh,  (long long)DHR_ * C_);
  cast(Wqr,  Wqrh,  (long long)NH_ * DHR_ * NLQ_);
  cast(Wuq,  Wuqh,  (long long)C_ * NLQ_);
  cast(Wo,   Woh,   (long long)C_ * C_);

  // WuvT[n][c] = Wuv[c][n]  (2048x512 -> 512x2048)
  transpose_cast<float><<<dim3(512 / 32, 2048 / 32, 1), blk, 0, stream>>>(
      Wuv, WuvT, 2048, 512, 512, 2048, 0, 0);
  // WukT[h][k][d] = Wuk[h*128+d][k]  (batched 128x512 -> 512x128)
  transpose_cast<float><<<dim3(512 / 32, 128 / 32, NH_), blk, 0, stream>>>(
      Wuk, WukT, 128, 512, 512, 128, 128LL * 512, 512LL * 128);

  // ---- projections ----
  gemmH(xh, Wdqh,  cqh,  4096, 512, 2048, 2048, 2048, 512, 0, 0, 0, 1);   // c_q
  gemmH(xh, Wdkvh, ckve, 4096, 512, 2048, 2048, 2048, DQK, 0, 0, 0, 1);   // c_kv
  gemmH(xh, Wkrh,  ckrh, 4096,  64, 2048, 2048, 2048,  64, 0, 0, 0, 1);   // c_kr
  {
    long long n = BT * (DHR_ / 2);
    rope_k_kernel<<<dim3((unsigned)((n + 255) / 256)), blk, 0, stream>>>(
        ckrh, cosp, sinp, ckve);                                          // k_r
  }
  // ckvT[b][k][s] = ckve[b][s][k]  (first 512 cols only)
  transpose_cast<half_t><<<dim3(512 / 32, 2048 / 32, B_), blk, 0, stream>>>(
      ckve, ckvT, T_, DV, DQK, T_, (long long)T_ * DQK, (long long)DV * T_);

  gemmH(cqh, Wqrh, cqrh, 4096, 1024, 512, 512, 512, 1024, 0, 0, 0, 1);    // c_qr
  {
    long long n = BT * NH_ * (DHR_ / 2);
    rope_q_kernel<<<dim3((unsigned)((n + 255) / 256)), blk, 0, stream>>>(
        cqrh, cosp, sinp, qkeh);                                          // q_r
  }

  // k_effT[h][k][q] = sum_d WukT[h][k][d] * Wuq[q*2048 + h*128 + d]
  gemmH(WukT, Wuqh, keffT, 512, 512, 128, 128, 2048, 512,
        512LL * 128, 128LL, 512LL * 512, NH_);

  // qk[b][h][t][k] = c_q[b] @ k_effT[h]^T -> qke[:,0:512]
  for (int b = 0; b < B_; ++b)
    gemmH(cqh + (long long)b * T_ * NLQ_, keffT,
          qkeh + (long long)b * NH_ * T_ * DQK,
          2048, 512, 512, 512, 512, DQK,
          0, 512LL * 512, (long long)T_ * DQK, NH_);

  // v_pre = W_uv^T @ W_o^T ;  v_effT = v_pre^T (2048 x 512)
  gemmH(WuvT, Woh, vpreh, 512, 2048, 2048, 2048, 2048, 2048, 0, 0, 0, 1);
  transpose_cast<half_t><<<dim3(2048 / 32, 512 / 32, 1), blk, 0, stream>>>(
      vpreh, veffT, 512, 2048, 2048, 512, 0, 0);

  // ---- flash attention core ----
  mla_flash<<<dim3(T_ / FT, B_ * NH_), blk, 0, stream>>>(
      qkeh, ckve, ckvT, ctxh, 1.0f / sqrtf((float)(HS_ + DHR_)));

  // ---- y[b][t][h*128+d] = ctx[b][h] @ v_effT[h]^T ----
  for (int b = 0; b < B_; ++b) {
    dim3 g(1, 2048 / BM, NH_);
    gemm_nt<false><<<g, blk, 0, stream>>>(
        ctxh + (long long)b * NH_ * T_ * DV, veffT,
        out + (long long)b * T_ * C_, nullptr,
        2048, 128, 512, 512, 512, 2048,
        (long long)T_ * DV, 128LL * 512, 128LL);
  }
}
